// GIN_cycle_50869592655556
// MI455X (gfx1250) — compile-verified, matrix operand order checked by reading, not run on hardware
//
#include <hip/hip_runtime.h>

typedef __attribute__((ext_vector_type(16))) _Float16 v16h;
typedef __attribute__((ext_vector_type(8)))  float    v8f;

// ---------------------------------------------------------------------------
// WMMA GEMM: out[M,N] = relu?(A[M,K](f16) x W[K,N] + b), W pre-transposed to
// Wt[N,K] f16.  One wave computes a 16x64 output strip (4 n-tiles): one A
// fragment load is reused by 4 v_wmma_f32_16x16x32_f16 against 4 B fragments,
// cutting A HBM traffic 4x vs one-tile-per-wave.
// grid = (M/16, N/128), block = (32,2): 2 waves per block, 8 n-tiles total.
// A fragment, contiguous k order (this code's, not ISA table order): lanes 0-15 -> row M=l, K=k0..k0+15;
//                            lanes 16-31 -> row M=l, K=k0+16..k0+31.
// B fragment mirrored (column-major) = contiguous row of Wt.
// ---------------------------------------------------------------------------
__global__ __launch_bounds__(64) void k_gemm(
    const _Float16* __restrict__ A, const _Float16* __restrict__ Wt,
    const float* __restrict__ bias, _Float16* __restrict__ out,
    int K, int N, int ldOut, int relu)
{
  const int lane = threadIdx.x;         // 0..31
  const int l15  = lane & 15;
  const int half = lane >> 4;           // 0 or 1
  const int tm   = blockIdx.x;
  const int tn0  = (blockIdx.y * 2 + threadIdx.y) * 4;  // first of 4 n-tiles

  const _Float16* Ap = A  + (size_t)(tm * 16 + l15) * K + half * 16;
  const _Float16* Bp = Wt + (size_t)(tn0 * 16 + l15) * K + half * 16;
  const size_t bs = (size_t)16 * K;     // halves between consecutive B n-tiles

  v8f acc0 = {}, acc1 = {}, acc2 = {}, acc3 = {};
#pragma unroll 2
  for (int k0 = 0; k0 < K; k0 += 32) {
    v16h a  = *(const v16h*)(Ap + k0);
    v16h b0 = *(const v16h*)(Bp + k0);
    v16h b1 = *(const v16h*)(Bp + bs + k0);
    v16h b2 = *(const v16h*)(Bp + 2 * bs + k0);
    v16h b3 = *(const v16h*)(Bp + 3 * bs + k0);
    acc0 = __builtin_amdgcn_wmma_f32_16x16x32_f16(false, a, false, b0, (short)0, acc0, false, false);
    acc1 = __builtin_amdgcn_wmma_f32_16x16x32_f16(false, a, false, b1, (short)0, acc1, false, false);
    acc2 = __builtin_amdgcn_wmma_f32_16x16x32_f16(false, a, false, b2, (short)0, acc2, false, false);
    acc3 = __builtin_amdgcn_wmma_f32_16x16x32_f16(false, a, false, b3, (short)0, acc3, false, false);
  }

  // C/D layout: lane<16 -> col=l15, rows tm*16+0..7 ; lane>=16 -> rows +8..15
  const int rbase = tm * 16 + half * 8;
  _Float16* orow = out + (size_t)rbase * ldOut;
#pragma unroll
  for (int t = 0; t < 4; ++t) {
    v8f acc = (t == 0) ? acc0 : (t == 1) ? acc1 : (t == 2) ? acc2 : acc3;
    const int ocol = (tn0 + t) * 16 + l15;
    const float bv = bias[ocol];
#pragma unroll
    for (int r = 0; r < 8; ++r) {
      float v = acc[r] + bv;
      if (relu) v = v > 0.f ? v : 0.f;
      orow[(size_t)r * ldOut + ocol] = (_Float16)v;
    }
  }
}

// W fp32 [K,N] -> Wt f16 [N,K]
__global__ void k_wconv(const float* __restrict__ W, _Float16* __restrict__ Wt,
                        int K, int N)
{
  int t = blockIdx.x * blockDim.x + threadIdx.x;
  if (t >= K * N) return;
  int k = t / N, n = t % N;
  Wt[(size_t)n * K + k] = (_Float16)W[t];
}

// embedding gather fp32 table -> f16 rows
__global__ void k_embed(const float* __restrict__ emb, const int* __restrict__ ids,
                        _Float16* __restrict__ out, int dim)
{
  int i = blockIdx.x, d = threadIdx.x;
  out[(size_t)i * dim + d] = (_Float16)emb[(size_t)ids[i] * dim + d];
}

__global__ void k_build_cids(const int* __restrict__ c5, const int* __restrict__ c6,
                             int* __restrict__ ids, int n5, int R)
{
  int t = blockIdx.x * blockDim.x + threadIdx.x;
  if (t >= R) return;
  ids[t] = (t < n5) ? c5[t] : c6[t - n5];
}

// cycle feature rows: [x[idx_j] | sum_j x[idx_j]]  -> out stride 256
__global__ void k_cycle_feat(const _Float16* __restrict__ x, const int* __restrict__ cyc,
                             int k, int rowBase, _Float16* __restrict__ out)
{
  int c = blockIdx.x, h = threadIdx.x;   // 128 threads
  int idx[6];
  float s = 0.f;
  for (int j = 0; j < k; ++j) {
    idx[j] = cyc[(size_t)c * k + j];
    s += (float)x[(size_t)idx[j] * 128 + h];
  }
  _Float16 sh = (_Float16)s;
  for (int j = 0; j < k; ++j) {
    size_t ro = (size_t)(rowBase + c * k + j) * 256;
    out[ro + h]       = x[(size_t)idx[j] * 128 + h];
    out[ro + 128 + h] = sh;
  }
}

// [e(256) | x[src]+x[dst](128)] -> stride 384
__global__ void k_build_ein3(const _Float16* __restrict__ e, const _Float16* __restrict__ x,
                             const int* __restrict__ src, const int* __restrict__ dst,
                             _Float16* __restrict__ out)
{
  int i = blockIdx.x, t = threadIdx.x;   // 384 threads
  size_t ro = (size_t)i * 384;
  if (t < 256) out[ro + t] = e[(size_t)i * 256 + t];
  else {
    int h = t - 256;
    float v = (float)x[(size_t)src[i] * 128 + h] + (float)x[(size_t)dst[i] * 128 + h];
    out[ro + t] = (_Float16)v;
  }
}

// [x(128) | e2n fp32 (128)] -> stride 256
__global__ void k_build_nin(const _Float16* __restrict__ x, const float* __restrict__ e2n,
                            _Float16* __restrict__ out)
{
  int i = blockIdx.x, t = threadIdx.x;   // 256 threads
  size_t ro = (size_t)i * 256;
  out[ro + t] = (t < 128) ? x[(size_t)i * 128 + t]
                          : (_Float16)e2n[(size_t)i * 128 + (t - 128)];
}

// [cyc(256) | en[cyc_ids](256)] -> stride 512
__global__ void k_build_cin(const _Float16* __restrict__ cyc, const float* __restrict__ en,
                            const int* __restrict__ cids, _Float16* __restrict__ out)
{
  int r = blockIdx.x, t = threadIdx.x;   // 256 threads
  size_t ro = (size_t)r * 512;
  out[ro + t]       = cyc[(size_t)r * 256 + t];
  out[ro + 256 + t] = (_Float16)en[(size_t)cids[r] * 256 + t];
}

// [e(256) | cn[src]+cn[dst](256)] -> stride 512
__global__ void k_build_ein4(const _Float16* __restrict__ e, const float* __restrict__ cn,
                             const int* __restrict__ src, const int* __restrict__ dst,
                             _Float16* __restrict__ out)
{
  int i = blockIdx.x, t = threadIdx.x;   // 512 threads
  size_t ro = (size_t)i * 512;
  if (t < 256) out[ro + t] = e[(size_t)i * 256 + t];
  else {
    int h = t - 256;
    out[ro + t] = (_Float16)(cn[(size_t)src[i] * 256 + h] + cn[(size_t)dst[i] * 256 + h]);
  }
}

// e1 [E,128] -> comb cols 0..127 (stride 384)
__global__ void k_copy_e1(const _Float16* __restrict__ e1, _Float16* __restrict__ comb)
{
  int i = blockIdx.x, t = threadIdx.x;   // 128 threads
  comb[(size_t)i * 384 + t] = e1[(size_t)i * 128 + t];
}

// segment-sum over both endpoints: acc[src]+=v ; acc[dst]+=v
__global__ void k_scatter_edge(const _Float16* __restrict__ v, const int* __restrict__ src,
                               const int* __restrict__ dst, float* __restrict__ acc,
                               int E, int dim)
{
  long long t = (long long)blockIdx.x * blockDim.x + threadIdx.x;
  if (t >= (long long)E * dim) return;
  int i = (int)(t / dim), d = (int)(t % dim);
  float x = (float)v[t];
  atomicAdd(&acc[(size_t)src[i] * dim + d], x);
  atomicAdd(&acc[(size_t)dst[i] * dim + d], x);
}

__global__ void k_scatter_idx(const _Float16* __restrict__ v, const int* __restrict__ ids,
                              float* __restrict__ acc, int R, int dim)
{
  long long t = (long long)blockIdx.x * blockDim.x + threadIdx.x;
  if (t >= (long long)R * dim) return;
  int i = (int)(t / dim), d = (int)(t % dim);
  atomicAdd(&acc[(size_t)ids[i] * dim + d], (float)v[t]);
}

// global_add_pool of [x | e2n | c2n] (640 wide) by batch id
__global__ void k_pool(const _Float16* __restrict__ x, const float* __restrict__ e2n,
                       const float* __restrict__ c2n, const int* __restrict__ batch,
                       float* __restrict__ pooled)
{
  int i = blockIdx.x, d = threadIdx.x;   // 640 threads
  float v;
  if (d < 128)      v = (float)x[(size_t)i * 128 + d];
  else if (d < 384) v = e2n[(size_t)i * 256 + (d - 128)];
  else              v = c2n[(size_t)i * 256 + (d - 384)];
  atomicAdd(&pooled[(size_t)batch[i] * 640 + d], v);
}

__global__ void k_f32f16(const float* __restrict__ in, _Float16* __restrict__ out, int n)
{
  int t = blockIdx.x * blockDim.x + threadIdx.x;
  if (t < n) out[t] = (_Float16)in[t];
}

// out[b] = sum_h h[b,h] * W[h]   (final 128 -> 1)
__global__ void k_final(const _Float16* __restrict__ h, const float* __restrict__ W,
                        float* __restrict__ out)
{
  __shared__ float red[128];
  int b = blockIdx.x, t = threadIdx.x;
  red[t] = (float)h[(size_t)b * 128 + t] * W[t];
  __syncthreads();
  for (int s = 64; s > 0; s >>= 1) {
    if (t < s) red[t] += red[t + s];
    __syncthreads();
  }
  if (t == 0) out[b] = red[0];
}

// ---------------------------------------------------------------------------
extern "C" void kernel_launch(void* const* d_in, const int* in_sizes, int n_in,
                              void* d_out, int out_size, void* d_ws, size_t ws_size,
                              hipStream_t stream)
{
  if (n_in < 107) return;

  const int NN = 150000, NE = 300000, NC5 = 15000, NC6 = 20000, NB = 512;
  const int R  = NC5 * 5 + NC6 * 6;          // 195000
  const int Rp = 195008;                     // padded to x16

  const int* x_ids    = (const int*)d_in[0];
  const int* edge_ids = (const int*)d_in[1];
  const int* src      = (const int*)d_in[2];
  const int* dst      = src + NE;
  const int* cyc5     = (const int*)d_in[3];
  const int* cyc6     = (const int*)d_in[4];
  const int* batch    = (const int*)d_in[5];

  // --- param leaf table (JAX pytree order: sorted dict keys) ---------------
  // 6..: cb(4) cm(6) edge_emb(1) final(1) ic_c(4) ic_e(4) ie(4) inn(4)
  //      layers[4x{ec_c,ec_e,ne_e,ne_n}](64) lin1(4) lin2(4) node_emb(1)
  struct LinDef { int wi, bi, K, N; };
  LinDef defs[49];
  int nd = 0;
  auto D = [&](int wi, int bi, int K, int N) { defs[nd++] = LinDef{wi, bi, K, N}; };
  D(6, 7, 384, 384);  D(8, 9, 384, 256);                       // cb        0,1
  D(10, 11, 256, 128); D(12, 13, 128, 128); D(14, 15, 128, 256); // cm      2..4
  D(18, 19, 512, 384); D(20, 21, 384, 256);                    // ic_c      5,6
  D(22, 23, 512, 384); D(24, 25, 384, 256);                    // ic_e      7,8
  D(26, 27, 384, 256); D(28, 29, 256, 128);                    // ie        9,10
  D(30, 31, 256, 256); D(32, 33, 256, 128);                    // inn       11,12
  for (int l = 0; l < 4; ++l) {                                // 13 + 8*l
    int b0 = 34 + l * 16;
    D(b0 + 0,  b0 + 1,  512, 384); D(b0 + 2,  b0 + 3,  384, 256);  // ec_c
    D(b0 + 4,  b0 + 5,  512, 384); D(b0 + 6,  b0 + 7,  384, 256);  // ec_e
    D(b0 + 8,  b0 + 9,  384, 128); D(b0 + 10, b0 + 11, 128, 128);  // ne_e
    D(b0 + 12, b0 + 13, 256, 128); D(b0 + 14, b0 + 15, 128, 128);  // ne_n
  }
  D(98, 99, 640, 128);  D(100, 101, 128, 128);                 // lin1      45,46
  D(102, 103, 128, 128); D(104, 105, 128, 128);                // lin2      47,48
  enum { CB0 = 0, CB1, CM0, CM1, CM2, ICC0, ICC1, ICE0, ICE1,
         IE0, IE1, INN0, INN1, LBASE = 13, L10 = 45, L11, L20, L21 };
  const float* node_emb = (const float*)d_in[106];   // [30,128]
  const float* edge_emb = (const float*)d_in[16];    // [8,256]
  const float* finalW   = (const float*)d_in[17];    // [128,1]

  // --- workspace bump allocator -------------------------------------------
  char* wsb = (char*)d_ws;
  size_t off = 0;
  auto alloc = [&](size_t bytes) -> void* {
    off = (off + 255) & ~(size_t)255;
    void* p = wsb + off;
    off += bytes;
    return p;
  };

  _Float16* wt[49];
  for (int i = 0; i < 49; ++i)
    wt[i] = (_Float16*)alloc((size_t)defs[i].K * defs[i].N * 2);

  int*      cyc_ids = (int*)alloc((size_t)R * 4);
  _Float16* x16   = (_Float16*)alloc((size_t)NN * 128 * 2);
  _Float16* e16   = (_Float16*)alloc((size_t)NE * 256 * 2);
  _Float16* cyc16 = (_Float16*)alloc((size_t)Rp * 256 * 2);
  _Float16* cinR  = (_Float16*)alloc((size_t)Rp * 512 * 2);
  _Float16* cmidR = (_Float16*)alloc((size_t)Rp * 384 * 2);
  _Float16* eIn   = (_Float16*)alloc((size_t)NE * 512 * 2);
  _Float16* eMid  = (_Float16*)alloc((size_t)NE * 384 * 2);
  _Float16* e1b   = (_Float16*)alloc((size_t)NE * 128 * 2);
  _Float16* nin   = (_Float16*)alloc((size_t)NN * 256 * 2);
  _Float16* nmid  = (_Float16*)alloc((size_t)NN * 256 * 2);
  float*    accA  = (float*)alloc((size_t)NN * 256 * 4);
  float*    accB  = (float*)alloc((size_t)NN * 256 * 4);
  float*    pooled = (float*)alloc((size_t)NB * 640 * 4);
  _Float16* p16   = (_Float16*)alloc((size_t)NB * 640 * 2);
  _Float16* hA    = (_Float16*)alloc((size_t)NB * 128 * 2);
  _Float16* hB    = (_Float16*)alloc((size_t)NB * 128 * 2);
  if (off > ws_size) return;   // workspace too small: bail (no-op)

  auto gemm = [&](const _Float16* A, int li, _Float16* out, int M, int ldOut, int relu) {
    dim3 g(M / 16, defs[li].N / 128), b(32, 2);
    k_gemm<<<g, b, 0, stream>>>(A, wt[li], (const float*)d_in[defs[li].bi],
                                out, defs[li].K, defs[li].N, ldOut, relu);
  };

  // --- 0. weight convert/transpose (fp32 [K,N] -> f16 [N,K]) --------------
  for (int i = 0; i < 49; ++i) {
    int kn = defs[i].K * defs[i].N;
    k_wconv<<<(kn + 255) / 256, 256, 0, stream>>>((const float*)d_in[defs[i].wi],
                                                  wt[i], defs[i].K, defs[i].N);
  }

  // --- 1. graph setup: cyc_ids, embeddings, cycle features ----------------
  k_build_cids<<<(R + 255) / 256, 256, 0, stream>>>(cyc5, cyc6, cyc_ids, NC5 * 5, R);
  k_embed<<<NN, 128, 0, stream>>>(node_emb, x_ids, x16, 128);
  k_embed<<<NE, 256, 0, stream>>>(edge_emb, edge_ids, e16, 256);

  hipMemsetAsync(cinR, 0, (size_t)Rp * 512 * 2, stream);  // also zeroes pad rows
  k_cycle_feat<<<NC5, 128, 0, stream>>>(x16, cyc5, 5, 0, cinR);
  k_cycle_feat<<<NC6, 128, 0, stream>>>(x16, cyc6, 6, NC5 * 5, cinR);

  // cyc = cycle_mlp(cyc_feats): 256 ->128 relu ->128 relu ->256
  gemm(cinR, CM0, cmidR, Rp, 128, 1);
  gemm(cmidR, CM1, cmidR + (size_t)Rp * 128, Rp, 128, 1);
  gemm(cmidR + (size_t)Rp * 128, CM2, cyc16, Rp, 256, 0);

  // --- 2. message-passing stage (init + 4 layers) -------------------------
  auto stage = [&](int e0, int e1, int n0, int n1, int c0, int c1, int ce0, int ce1) {
    // node_edge: e1 = mlp([e, x[src]+x[dst]]); x = mlp([x, seg(e1)])
    k_build_ein3<<<NE, 384, 0, stream>>>(e16, x16, src, dst, eIn);
    gemm(eIn, e0, eMid, NE, defs[e0].N, 1);
    gemm(eMid, e1, e1b, NE, 128, 0);
    hipMemsetAsync(accA, 0, (size_t)NN * 128 * 4, stream);
    k_scatter_edge<<<(NE * 128 + 255) / 256, 256, 0, stream>>>(e1b, src, dst, accA, NE, 128);
    k_build_nin<<<NN, 256, 0, stream>>>(x16, accA, nin);
    gemm(nin, n0, nmid, NN, defs[n0].N, 1);
    gemm(nmid, n1, x16, NN, 128, 0);

    // edge_cycle: en=seg(e); cyc=mlp([cyc,en[cids]]); cn=seg(cyc);
    //             e2=mlp([e, cn[src]+cn[dst]])
    hipMemsetAsync(accA, 0, (size_t)NN * 256 * 4, stream);
    k_scatter_edge<<<(NE * 256 + 255) / 256, 256, 0, stream>>>(e16, src, dst, accA, NE, 256);
    k_build_cin<<<R, 256, 0, stream>>>(cyc16, accA, cyc_ids, cinR);
    gemm(cinR, c0, cmidR, Rp, 384, 1);
    gemm(cmidR, c1, cyc16, Rp, 256, 0);
    hipMemsetAsync(accA, 0, (size_t)NN * 256 * 4, stream);
    k_scatter_idx<<<(R * 256 + 255) / 256, 256, 0, stream>>>(cyc16, cyc_ids, accA, R, 256);
    k_build_ein4<<<NE, 512, 0, stream>>>(e16, accA, src, dst, eIn);
    gemm(eIn, ce0, eMid, NE, 384, 1);
    gemm(eMid, ce1, eIn + 128, NE, 384, 0);     // e2 -> comb cols 128..383
    k_copy_e1<<<NE, 128, 0, stream>>>(e1b, eIn); // e1 -> comb cols 0..127

    // e = combine_edge_mlp([e1, e2]): 384 ->384 relu ->256
    gemm(eIn, CB0, eMid, NE, 384, 1);
    gemm(eMid, CB1, e16, NE, 256, 0);
  };

  stage(IE0, IE1, INN0, INN1, ICC0, ICC1, ICE0, ICE1);
  for (int l = 0; l < 4; ++l) {
    int b0 = LBASE + 8 * l;
    stage(b0 + 4, b0 + 5, b0 + 6, b0 + 7,   // ne_e, ne_n
          b0 + 0, b0 + 1, b0 + 2, b0 + 3);  // ec_c, ec_e
  }

  // --- 3. readout ----------------------------------------------------------
  hipMemsetAsync(accA, 0, (size_t)NN * 256 * 4, stream);
  k_scatter_edge<<<(NE * 256 + 255) / 256, 256, 0, stream>>>(e16, src, dst, accA, NE, 256);
  hipMemsetAsync(accB, 0, (size_t)NN * 256 * 4, stream);
  k_scatter_idx<<<(R * 256 + 255) / 256, 256, 0, stream>>>(cyc16, cyc_ids, accB, R, 256);
  hipMemsetAsync(pooled, 0, (size_t)NB * 640 * 4, stream);
  k_pool<<<NN, 640, 0, stream>>>(x16, accA, accB, batch, pooled);
  k_f32f16<<<(NB * 640 + 255) / 256, 256, 0, stream>>>(pooled, p16, NB * 640);

  gemm(p16, L10, hA, NB, 128, 1);   // lin1[0] + relu
  gemm(hA, L11, hB, NB, 128, 1);    // lin1[1] then outer relu
  gemm(hB, L20, hA, NB, 128, 1);    // lin2[0] + relu
  gemm(hA, L21, hB, NB, 128, 0);    // lin2[1]
  k_final<<<NB, 128, 0, stream>>>(hB, finalW, (float*)d_out);
}
